// MultiheadAttention_9062380994635
// MI455X (gfx1250) — matrix operand order repaired and checked by reading, not compiled, not run on hardware
//
#include <hip/hip_runtime.h>

// ---------------- problem constants ----------------
#define B_    4
#define S_    2048
#define D_    512
#define E_    512
#define H_    8
#define HD_   64
#define T3E   1536          // 3*E
#define NTOK  8192          // B*S
#define ALPHA_ 1.0f

typedef _Float16 h16 __attribute__((ext_vector_type(16)));
typedef _Float16 h8  __attribute__((ext_vector_type(8)));
typedef float    f8  __attribute__((ext_vector_type(8)));

static __device__ __forceinline__ f8 wmma_f16(h16 a, h16 b, f8 c) {
  // v_wmma_f32_16x16x32_f16 : D = A(16x32 f16) x B(32x16 f16) + C(16x16 f32)
  return __builtin_amdgcn_wmma_f32_16x16x32_f16(false, a, false, b, (short)0, c, false, false);
}

static __device__ __forceinline__ h16 cat16(h8 lo, h8 hi) {
  return __builtin_shufflevector(lo, hi, 0,1,2,3,4,5,6,7,8,9,10,11,12,13,14,15);
}

// A-fragment 16x32 (MxK), source row-major [M x ldk].
// Lane (r, hf): row = rowBase + r ; k = k0 + hf*8 + {0..7}  and  k0 + 16 + hf*8 + {0..7}
static __device__ __forceinline__ h16 load_a_frag(const _Float16* p, int ld, int row, int k0, int hf) {
  const _Float16* base = p + (size_t)row * ld + k0 + hf * 8;
  h8 lo = *(const h8*)(base);
  h8 hi = *(const h8*)(base + 16);
  return cat16(lo, hi);
}

// B-fragment 32x16 (KxN) loaded from a row-major [N x ldk] source (i.e. B^T rows).
// Lane (r, hf): n = r ; k = k0 + hf*8 + {0..7}  and  k0 + 16 + hf*8 + {0..7}
static __device__ __forceinline__ h16 load_bt_frag(const _Float16* p, int ld, int nrow, int k0, int hf) {
  const _Float16* base = p + (size_t)nrow * ld + k0 + hf * 8;
  h8 lo = *(const h8*)(base);
  h8 hi = *(const h8*)(base + 16);
  return cat16(lo, hi);
}

// ---------------- tiny helper kernels ----------------
__global__ void k_init(float* fro, unsigned* xmax) { *fro = 0.0f; *xmax = 0u; }

__global__ void k_f2h(const float* __restrict__ src, _Float16* __restrict__ dst, int n) {
  int i = blockIdx.x * blockDim.x + threadIdx.x;
  if (i < n) dst[i] = (_Float16)src[i];
}

// convert x -> f16 and compute max over tokens of ||x_token||^2 (atomicMax on float bits)
__global__ __launch_bounds__(256) void k_xprep(const float* __restrict__ x,
                                               _Float16* __restrict__ xh,
                                               unsigned* __restrict__ xmax_bits) {
  int row = blockIdx.x;               // one token per block, 512 elems / 256 threads
  int t = threadIdx.x;
  const float* xr = x + (size_t)row * D_;
  float a0 = xr[2 * t], a1 = xr[2 * t + 1];
  xh[(size_t)row * D_ + 2 * t]     = (_Float16)a0;
  xh[(size_t)row * D_ + 2 * t + 1] = (_Float16)a1;
  float ss = a0 * a0 + a1 * a1;
#pragma unroll
  for (int m = 16; m >= 1; m >>= 1) ss += __shfl_xor(ss, m, 32);
  __shared__ float sbuf[8];
  if ((t & 31) == 0) sbuf[t >> 5] = ss;
  __syncthreads();
  if (t == 0) {
    float s = 0.0f;
#pragma unroll
    for (int i = 0; i < 8; ++i) s += sbuf[i];
    atomicMax(xmax_bits, __float_as_uint(s));  // s >= 0 -> bit pattern is order-preserving
  }
}

// ---------------- QKV projection GEMM ----------------
// qkv[8192 x 1536] = xh[8192 x 512] @ Wqkv^T + b.
// Epilogue: f16 qkv store; V channels ALSO stored transposed (vth[b,h,d,s]) so the
// attention P@V B-fragments become contiguous b128 loads; Sum(q^2) accumulated for fro_q.
// One wave computes a 16x64 output tile. waves = 512*24 = 12288 -> 1536 blocks of 8 waves.
__global__ __launch_bounds__(256) void k_qkv_gemm(const _Float16* __restrict__ xh,
                                                  const _Float16* __restrict__ wq,
                                                  const float* __restrict__ bias,
                                                  _Float16* __restrict__ qkvh,
                                                  _Float16* __restrict__ vth,
                                                  float* __restrict__ fro_acc) {
  int lane = threadIdx.x & 31, wid = threadIdx.x >> 5;
  int r = lane & 15, hf = lane >> 4;
  int w = blockIdx.x * 8 + wid;
  int rowBlk = w / 24, colBlk = w % 24;
  int mBase = rowBlk * 16, nBase = colBlk * 64;

  f8 acc[4] = {};
  for (int k0 = 0; k0 < D_; k0 += 32) {
    if (k0 + 32 < D_) {
      __builtin_prefetch(xh + (size_t)(mBase + r) * D_ + k0 + 32, 0, 3);
      __builtin_prefetch(wq + (size_t)(nBase + r) * D_ + k0 + 32, 0, 3);
    }
    h16 af = load_a_frag(xh, D_, mBase + r, k0, hf);
#pragma unroll
    for (int s = 0; s < 4; ++s) {
      h16 bf = load_bt_frag(wq, D_, nBase + 16 * s + r, k0, hf);
      acc[s] = wmma_f16(af, bf, acc[s]);
    }
  }
  float qs = 0.0f;
#pragma unroll
  for (int s = 0; s < 4; ++s) {
    int n = nBase + 16 * s + r;
    float bv = bias[n];
    int hh = n / 192, off = n % 192;      // head-local channel: [0,64)=q [64,128)=k [128,192)=v
    bool isq = off < 64;
    bool isv = off >= 128;
    _Float16* vtr = vth + ((size_t)hh * HD_ + (off - 128)) * S_;  // row for this v-channel (batch added below)
#pragma unroll
    for (int v = 0; v < 8; ++v) {
      float val = acc[s][v] + bv;
      int m = mBase + v + 8 * hf;
      qkvh[(size_t)m * T3E + n] = (_Float16)val;
      if (isq) qs += val * val;
      if (isv) {
        int bb = m >> 11, ss2 = m & (S_ - 1);          // m = b*S + s
        vtr[((size_t)bb * H_ * HD_) * S_ + ss2] = (_Float16)val;
      }
    }
  }
#pragma unroll
  for (int m = 16; m >= 1; m >>= 1) qs += __shfl_xor(qs, m, 32);
  if (lane == 0) atomicAdd(fro_acc, qs);
}

// ---------------- key-norm precompute ----------------
// kn2[(b*H+h)*S + s] = ||k_{b,h,s}||^2 ; computed once instead of per query-block wave.
__global__ __launch_bounds__(256) void k_knorm(const _Float16* __restrict__ qkvh,
                                               float* __restrict__ kn2) {
  int idx = blockIdx.x * blockDim.x + threadIdx.x;   // (b*H+h)*S + s, 65536 total
  int s  = idx & (S_ - 1);
  int bh = idx >> 11;
  int b = bh >> 3, h = bh & 7;
  const _Float16* kr = qkvh + (size_t)(b * S_ + s) * T3E + h * (3 * HD_) + HD_;
  float ss = 0.0f;
#pragma unroll
  for (int i = 0; i < HD_; i += 8) {
    h8 v = *(const h8*)(kr + i);
#pragma unroll
    for (int j = 0; j < 8; ++j) { float f = (float)v[j]; ss += f * f; }
  }
  kn2[idx] = ss;
}

// ---------------- L2-distance flash attention ----------------
// exponent(i,j) = c*(2*q.k - ||k_j||^2), c = ALPHA/(fro_q * inf_2_x)  (||q_i||^2 cancels in softmax)
// One wave per (b,h,16-query block); 32 keys per iteration; online softmax.
__global__ __launch_bounds__(256) void k_attn(const _Float16* __restrict__ qkvh,
                                              const _Float16* __restrict__ vth,
                                              const float* __restrict__ kn2,
                                              _Float16* __restrict__ ctxh,
                                              const float* __restrict__ fro_acc,
                                              const unsigned* __restrict__ xmax_bits) {
  __shared__ __align__(16) _Float16 ldsP[8][16 * 32];   // per-wave P tile
  int lane = threadIdx.x & 31, wid = threadIdx.x >> 5;
  int r = lane & 15, hf = lane >> 4;
  int w = blockIdx.x * 8 + wid;                  // 4096 waves total
  int qb = w & 127;
  int h  = (w >> 7) & 7;
  int b  = w >> 10;

  const _Float16* base = qkvh + (size_t)b * S_ * T3E + h * (3 * HD_);
  const _Float16* qp  = base;
  const _Float16* kp  = base + HD_;
  const _Float16* vtp = vth + (size_t)(b * H_ + h) * HD_ * S_;   // [d][s], ld = S_
  const float*    knp = kn2 + (size_t)(b * H_ + h) * S_;

  float c  = ALPHA_ / (sqrtf(*fro_acc) * sqrtf(__uint_as_float(*xmax_bits)));
  float c2 = 2.0f * c;

  h16 aq0 = load_a_frag(qp, T3E, qb * 16 + r, 0, hf);
  h16 aq1 = load_a_frag(qp, T3E, qb * 16 + r, 32, hf);

  f8 o[4] = {};
  float rowm[8], rowl[8];
#pragma unroll
  for (int v = 0; v < 8; ++v) { rowm[v] = -__builtin_inff(); rowl[v] = 0.0f; }
  _Float16* pl = ldsP[wid];

  for (int j0 = 0; j0 < S_; j0 += 32) {
    if (j0 + 32 < S_) {
      __builtin_prefetch(kp + (size_t)(j0 + 32 + r) * T3E, 0, 3);
      __builtin_prefetch(kp + (size_t)(j0 + 48 + r) * T3E, 0, 3);
    }
    f8 st[2]; float kn[2];
#pragma unroll
    for (int t = 0; t < 2; ++t) {
      int krow = j0 + 16 * t + r;
      h16 bk0 = load_bt_frag(kp, T3E, krow, 0, hf);
      h16 bk1 = load_bt_frag(kp, T3E, krow, 32, hf);
      f8 s = {};
      s = wmma_f16(aq0, bk0, s);
      s = wmma_f16(aq1, bk1, s);
      st[t] = s;
      kn[t] = c * knp[krow];     // precomputed ||k_j||^2 for this lane's column
    }
    float e0[8], e1[8], tmax[8];
#pragma unroll
    for (int v = 0; v < 8; ++v) {
      e0[v] = c2 * st[0][v] - kn[0];
      e1[v] = c2 * st[1][v] - kn[1];
      tmax[v] = fmaxf(e0[v], e1[v]);
    }
#pragma unroll
    for (int m = 8; m >= 1; m >>= 1)          // row reduce within each 16-lane half
#pragma unroll
      for (int v = 0; v < 8; ++v) tmax[v] = fmaxf(tmax[v], __shfl_xor(tmax[v], m, 32));
    float p0[8], p1[8], rs[8];
#pragma unroll
    for (int v = 0; v < 8; ++v) {
      float mn = fmaxf(rowm[v], tmax[v]);
      float sc = __expf(rowm[v] - mn);        // first iter: exp(-inf)=0
      rowm[v] = mn;
      p0[v] = __expf(e0[v] - mn);
      p1[v] = __expf(e1[v] - mn);
      rs[v] = p0[v] + p1[v];
      rowl[v] *= sc;
#pragma unroll
      for (int nb = 0; nb < 4; ++nb) o[nb][v] *= sc;
    }
#pragma unroll
    for (int m = 8; m >= 1; m >>= 1)
#pragma unroll
      for (int v = 0; v < 8; ++v) rs[v] += __shfl_xor(rs[v], m, 32);
#pragma unroll
    for (int v = 0; v < 8; ++v) rowl[v] += rs[v];

    // stage P (16x32 f16, row-major) in per-wave LDS, reload as A-fragment
#pragma unroll
    for (int v = 0; v < 8; ++v) {
      int m = v + 8 * hf;
      pl[m * 32 + r]      = (_Float16)p0[v];
      pl[m * 32 + 16 + r] = (_Float16)p1[v];
    }
    h16 ap = load_a_frag(pl, 32, r, 0, hf);   // same-wave LDS RAW; compiler inserts s_wait_dscnt
#pragma unroll
    for (int nb = 0; nb < 4; ++nb) {
      // V^T layout: contiguous b128 loads (no more 2-byte gathers)
      h16 bv = load_bt_frag(vtp, S_, nb * 16 + r, j0, hf);
      o[nb] = wmma_f16(ap, bv, o[nb]);
    }
  }

  int tokBase = b * S_ + qb * 16;
#pragma unroll
  for (int nb = 0; nb < 4; ++nb)
#pragma unroll
    for (int v = 0; v < 8; ++v) {
      float val = o[nb][v] / rowl[v];
      int m = v + 8 * hf;
      ctxh[(size_t)(tokBase + m) * E_ + h * HD_ + nb * 16 + r] = (_Float16)val;
    }
}

// ---------------- output projection GEMM ----------------
// out[8192 x 512] = ctx[8192 x 512] @ Wo^T + bo, f32 output. 4096 waves -> 512 blocks.
__global__ __launch_bounds__(256) void k_out_gemm(const _Float16* __restrict__ ah,
                                                  const _Float16* __restrict__ wh,
                                                  const float* __restrict__ bias,
                                                  float* __restrict__ out) {
  int lane = threadIdx.x & 31, wid = threadIdx.x >> 5;
  int r = lane & 15, hf = lane >> 4;
  int w = blockIdx.x * 8 + wid;
  int rowBlk = w >> 3, colBlk = w & 7;
  int mBase = rowBlk * 16, nBase = colBlk * 64;

  f8 acc[4] = {};
  for (int k0 = 0; k0 < E_; k0 += 32) {
    if (k0 + 32 < E_) {
      __builtin_prefetch(ah + (size_t)(mBase + r) * E_ + k0 + 32, 0, 3);
      __builtin_prefetch(wh + (size_t)(nBase + r) * E_ + k0 + 32, 0, 3);
    }
    h16 af = load_a_frag(ah, E_, mBase + r, k0, hf);
#pragma unroll
    for (int s = 0; s < 4; ++s) {
      h16 bf = load_bt_frag(wh, E_, nBase + 16 * s + r, k0, hf);
      acc[s] = wmma_f16(af, bf, acc[s]);
    }
  }
#pragma unroll
  for (int s = 0; s < 4; ++s) {
    int n = nBase + 16 * s + r;
    float bv = bias[n];
#pragma unroll
    for (int v = 0; v < 8; ++v) {
      int m = mBase + v + 8 * hf;
      out[(size_t)m * E_ + n] = acc[s][v] + bv;
    }
  }
}

// ---------------- host launcher ----------------
extern "C" void kernel_launch(void* const* d_in, const int* in_sizes, int n_in,
                              void* d_out, int out_size, void* d_ws, size_t ws_size,
                              hipStream_t stream) {
  const float* x    = (const float*)d_in[0];   // [4,2048,512]
  const float* Wqkv = (const float*)d_in[1];   // [1536,512]
  const float* bqkv = (const float*)d_in[2];   // [1536]
  const float* Wo   = (const float*)d_in[3];   // [512,512]
  const float* bo   = (const float*)d_in[4];   // [512]
  float* out = (float*)d_out;                  // [4,2048,512]

  char* ws = (char*)d_ws;
  size_t off = 0;
  _Float16* xh   = (_Float16*)(ws + off); off += (size_t)NTOK * D_ * 2;        // 8 MB
  _Float16* wqh  = (_Float16*)(ws + off); off += (size_t)T3E * D_ * 2;         // 1.5 MB
  _Float16* woh  = (_Float16*)(ws + off); off += (size_t)E_ * E_ * 2;          // 0.5 MB
  _Float16* qkvh = (_Float16*)(ws + off); off += (size_t)NTOK * T3E * 2;       // 24 MB
  _Float16* ctxh = (_Float16*)(ws + off); off += (size_t)NTOK * E_ * 2;        // 8 MB
  _Float16* vth  = (_Float16*)(ws + off); off += (size_t)B_ * H_ * HD_ * S_ * 2; // 8 MB (V transposed)
  float*    kn2  = (float*)(ws + off);    off += (size_t)B_ * H_ * S_ * 4;     // 256 KB
  float*    fro  = (float*)(ws + off);
  unsigned* xmax = (unsigned*)(ws + off + 4);

  k_init<<<1, 1, 0, stream>>>(fro, xmax);
  k_f2h<<<(T3E * D_ + 255) / 256, 256, 0, stream>>>(Wqkv, wqh, T3E * D_);
  k_f2h<<<(E_ * E_ + 255) / 256, 256, 0, stream>>>(Wo, woh, E_ * E_);
  k_xprep<<<NTOK, 256, 0, stream>>>(x, xh, xmax);
  k_qkv_gemm<<<1536, 256, 0, stream>>>(xh, wqh, bqkv, qkvh, vth, fro);  // 12288 waves
  k_knorm<<<(B_ * H_ * S_) / 256, 256, 0, stream>>>(qkvh, kn2);
  k_attn<<<512, 256, 0, stream>>>(qkvh, vth, kn2, ctxh, fro, xmax);     // 4096 waves
  k_out_gemm<<<512, 256, 0, stream>>>(ctxh, woh, bo, out);              // 4096 waves
}
